// EnhancedAttention_84516366451013
// MI455X (gfx1250) — matrix operand order repaired and checked by reading, not compiled, not run on hardware
//
#include <hip/hip_runtime.h>
#include <hip/hip_bf16.h>

// ---------------------------------------------------------------------------
// EnhancedAttention (BigBird-style band+global sparse attention) for MI455X.
// B=2, S=2048, D=1024, H=16, HD=64, WINDOW=64 (band +/-32), GLOBAL=16.
// All matmuls via v_wmma_f32_16x16x32_bf16 (wave32). Dense attn output is
// written with non-temporal stores (537 MB stream dominates the roofline).
// ---------------------------------------------------------------------------

typedef __attribute__((ext_vector_type(16))) __bf16 v16bf;
typedef __attribute__((ext_vector_type(8)))  __bf16 v8bf;
typedef __attribute__((ext_vector_type(4)))  __bf16 v4bf;
typedef __attribute__((ext_vector_type(8)))  float  v8f;
typedef __attribute__((ext_vector_type(4)))  float  v4f;   // native vector for NT stores

#define WMMA_BF16(a, b, c) \
    __builtin_amdgcn_wmma_f32_16x16x32_bf16(false, (a), false, (b), (short)0, (c), false, false)

static constexpr int  kB   = 2;
static constexpr int  kS   = 2048;
static constexpr int  kD   = 1024;
static constexpr int  kH   = 16;
static constexpr int  kHD  = 64;
static constexpr int  kTok = kB * kS;                         // 4096 tokens
static constexpr long kAttnElems = (long)kB * kH * kS * kS;   // 134,217,728
static constexpr long kOutElems  = (long)kB * kS * kD;        // 4,194,304

// ---- fragment builders -----------------------------------------------------
// A-matrix 16x32 bf16 (contiguous k order used here, matched by B): lane L holds row M=L%16;
//   half h -> K = k0 + (L/16)*16 + h
__device__ __forceinline__ v16bf make_a_frag(const __bf16* p /* row base + hi*16 + k0 */) {
    v8bf lo = *(const v8bf*)p;
    v8bf hi = *(const v8bf*)(p + 8);
    return __builtin_shufflevector(lo, hi, 0,1,2,3,4,5,6,7,8,9,10,11,12,13,14,15);
}

// B-matrix 32x16 bf16: lane L holds column N=L%16; half h -> K = k0 + (L/16)*16 + h
__device__ __forceinline__ v16bf make_b_frag(const __bf16* p /* col base + hi*16 + k0 */) {
    v8bf lo = *(const v8bf*)p;
    v8bf hi = *(const v8bf*)(p + 8);
    return __builtin_shufflevector(lo, hi, 0,1,2,3,4,5,6,7,8,9,10,11,12,13,14,15);
}

// ---- kernel 1: f32 -> bf16 (vectorized 4-wide) ------------------------------
__global__ void cvt_f32_to_bf16_v4(const v4f* __restrict__ in, v4bf* __restrict__ out, int n4) {
    for (int i = blockIdx.x * blockDim.x + threadIdx.x; i < n4; i += gridDim.x * blockDim.x) {
        v4f f = in[i];
        v4bf o = {(__bf16)f.x, (__bf16)f.y, (__bf16)f.z, (__bf16)f.w};
        out[i] = o;
    }
}

// ---- kernel 2/5: C[m,n] = sum_k X[m,k]*W[n,k] + bias[n] --------------------
// mode 0: store bf16, head-split token-major   [B,H,S,HD]   (Q, K)
// mode 1: store f32,  flat                     [M,N]        (final out)
// mode 2: store bf16, head-split d-major       [B,H,HD,S]   (V, for ctx B-frags)
__global__ __launch_bounds__(256) void gemm_xwT(const __bf16* __restrict__ X,
                                                const __bf16* __restrict__ W,
                                                const float*  __restrict__ bias,
                                                void* __restrict__ out,
                                                int M, int N, int K, int mode) {
    const int lane  = threadIdx.x & 31;
    const int ln    = lane & 15;
    const int hi    = lane >> 4;
    const int tilesN = N >> 4;
    const int wg = blockIdx.x * (blockDim.x >> 5) + (threadIdx.x >> 5);
    const int tm = wg / tilesN, tn = wg % tilesN;
    if (tm >= (M >> 4)) return;
    const int m0 = tm << 4, n0 = tn << 4;

    const __bf16* arow = X + (size_t)(m0 + ln) * K + hi * 16;
    const __bf16* brow = W + (size_t)(n0 + ln) * K + hi * 16;

    v8f c = {};
#pragma unroll 4
    for (int k0 = 0; k0 < K; k0 += 32) {
        v16bf a = make_a_frag(arow + k0);
        v16bf b = make_b_frag(brow + k0);
        c = WMMA_BF16(a, b, c);
    }

    const int   col = n0 + ln;
    const float bn  = bias[col];
    if (mode == 0) {          // bf16 token-major [B,H,S,HD]
        __bf16* O = (__bf16*)out;
        const int head = col >> 6, dc = col & 63;
#pragma unroll
        for (int r = 0; r < 8; ++r) {
            int t = m0 + r + 8 * hi, bb = t >> 11, s = t & 2047;
            O[(((size_t)bb * kH + head) * kS + s) * kHD + dc] = (__bf16)(c[r] + bn);
        }
    } else if (mode == 2) {   // bf16 d-major [B,H,HD,S]
        __bf16* O = (__bf16*)out;
        const int head = col >> 6, dc = col & 63;
#pragma unroll
        for (int r = 0; r < 8; ++r) {
            int t = m0 + r + 8 * hi, bb = t >> 11, s = t & 2047;
            O[(((size_t)bb * kH + head) * kHD + dc) * kS + s] = (__bf16)(c[r] + bn);
        }
    } else {                  // f32 flat [M,N]
        float* O = (float*)out;
#pragma unroll
        for (int r = 0; r < 8; ++r)
            O[(size_t)(m0 + r + 8 * hi) * N + col] = c[r] + bn;
    }
}

// ---- kernel 3: zero-fill attn region (streaming, NT stores) ----------------
__global__ void zero_f32v4(v4f* __restrict__ p, long n4) {
    const v4f z = {0.f, 0.f, 0.f, 0.f};
    for (long i = blockIdx.x * (long)blockDim.x + threadIdx.x; i < n4;
         i += (long)gridDim.x * blockDim.x)
        __builtin_nontemporal_store(z, p + i);
}

// ---- kernel 4: sparse attention --------------------------------------------
// One wave per (b*h, 16-row query block). Active column tiles:
//   qb==0 : all 128 tiles (rows < GLOBAL are dense)
//   else  : tile 0 (global cols) + band tiles [qb-2, qb+2] clipped
__global__ __launch_bounds__(128) void attn_sparse(const __bf16* __restrict__ Q,
                                                   const __bf16* __restrict__ Kk,
                                                   const __bf16* __restrict__ Vt,
                                                   float* __restrict__ attn,
                                                   __bf16* __restrict__ ctx) {
    __shared__ __bf16 lds_p[4][16 * 32];     // per-wave 16x32 probability pair buffer

    const int lane = threadIdx.x & 31;
    const int ln   = lane & 15;
    const int hi   = lane >> 4;
    const int wslot = threadIdx.x >> 5;
    const int wg = blockIdx.x * (blockDim.x >> 5) + wslot;      // 0..4095
    const int bh = wg >> 7;                  // b*16+h
    const int qb = wg & 127;                 // query block
    const int q0 = qb << 4;

    const __bf16* Qbh = Q  + (size_t)bh * kS * kHD;
    const __bf16* Kbh = Kk + (size_t)bh * kS * kHD;
    const __bf16* Vbh = Vt + (size_t)bh * kHD * kS;   // d-major [HD][S]

    // active tile list
    int lo = 0, extra = 0, nt = 128;
    if (qb != 0) {
        lo = qb - 2 < 0 ? 0 : qb - 2;
        int hib = qb + 2 > 127 ? 127 : qb + 2;
        extra = (lo > 0) ? 1 : 0;
        nt = hib - lo + 1 + extra;
    }
    auto tile_of = [&](int t) { return extra ? (t == 0 ? 0 : lo + t - 1) : lo + t; };

    // Q fragments (K = 64 -> two 16x16x32 steps)
    const __bf16* qp = Qbh + (size_t)(q0 + ln) * kHD + hi * 16;
    const v16bf a0 = make_a_frag(qp);
    const v16bf a1 = make_a_frag(qp + 32);

    auto score16 = [&](int j0) -> v8f {
        const __bf16* kp = Kbh + (size_t)(j0 + ln) * kHD + hi * 16;
        v16bf b0 = make_b_frag(kp);
        v16bf b1 = make_b_frag(kp + 32);
        v8f c = {};
        c = WMMA_BF16(a0, b0, c);
        c = WMMA_BF16(a1, b1, c);
        return c;
    };
    auto masked = [&](float s, int r, int j0) -> float {
        int i = q0 + r + 8 * hi;
        int j = j0 + ln;
        bool act = (i < 16) | (j < 16) | ((i - j <= 32) & (j - i <= 32));
        return act ? s * 0.125f : -1e9f;
    };

    // pass 1: row max (rows live across 16-lane halves -> shfl_xor reduce)
    float m8[8];
#pragma unroll
    for (int r = 0; r < 8; ++r) m8[r] = -3.0e38f;
    for (int t = 0; t < nt; ++t) {
        int j0 = tile_of(t) << 4;
        v8f c = score16(j0);
#pragma unroll
        for (int r = 0; r < 8; ++r) m8[r] = fmaxf(m8[r], masked(c[r], r, j0));
    }
#pragma unroll
    for (int off = 1; off < 16; off <<= 1)
#pragma unroll
        for (int r = 0; r < 8; ++r) m8[r] = fmaxf(m8[r], __shfl_xor(m8[r], off, 32));

    // pass 2: row sum
    float s8[8];
#pragma unroll
    for (int r = 0; r < 8; ++r) s8[r] = 0.f;
    for (int t = 0; t < nt; ++t) {
        int j0 = tile_of(t) << 4;
        v8f c = score16(j0);
#pragma unroll
        for (int r = 0; r < 8; ++r) s8[r] += __expf(masked(c[r], r, j0) - m8[r]);
    }
#pragma unroll
    for (int off = 1; off < 16; off <<= 1)
#pragma unroll
        for (int r = 0; r < 8; ++r) s8[r] += __shfl_xor(s8[r], off, 32);
    float inv8[8];
#pragma unroll
    for (int r = 0; r < 8; ++r) inv8[r] = 1.f / s8[r];

    // pass 3: normalize, write attn (NT), accumulate ctx = P @ V (paired tiles, K=32)
    __bf16* lw = &lds_p[wslot][0];
    v8f acc[4] = {{}, {}, {}, {}};
    const int npairs = (nt + 1) >> 1;
    for (int pr = 0; pr < npairs; ++pr) {
        const int  t0 = 2 * pr;
        const bool has1 = (t0 + 1) < nt;
        const int  j0 = tile_of(t0) << 4;
        const int  j1 = has1 ? (tile_of(t0 + 1) << 4) : j0;

        {   // tile 0 -> LDS cols 0..15
            v8f c = score16(j0);
#pragma unroll
            for (int r = 0; r < 8; ++r) {
                int i = q0 + r + 8 * hi, j = j0 + ln;
                float p = __expf(masked(c[r], r, j0) - m8[r]) * inv8[r];
                __builtin_nontemporal_store(p, attn + ((size_t)bh * kS + i) * kS + j);
                lw[(r + 8 * hi) * 32 + ln] = (__bf16)p;
            }
        }
        if (has1) {   // tile 1 -> LDS cols 16..31
            v8f c = score16(j1);
#pragma unroll
            for (int r = 0; r < 8; ++r) {
                int i = q0 + r + 8 * hi, j = j1 + ln;
                float p = __expf(masked(c[r], r, j1) - m8[r]) * inv8[r];
                __builtin_nontemporal_store(p, attn + ((size_t)bh * kS + i) * kS + j);
                lw[(r + 8 * hi) * 32 + 16 + ln] = (__bf16)p;
            }
        } else {      // zero-pad odd pair
#pragma unroll
            for (int r = 0; r < 8; ++r)
                lw[(r + 8 * hi) * 32 + 16 + ln] = (__bf16)0.f;
        }

        asm volatile("s_wait_dscnt 0" ::: "memory");   // LDS RAW fence (CDNA5 split counter)

        // re-fragment P (16x32) from LDS into A layout
        v16bf af = make_a_frag(lw + ln * 32 + hi * 16);

        // V B-fragments from d-major layout: contiguous 16 tokens per half-wave
        const int jb = hi ? j1 : j0;   // K rows 0..15 <- tile0 tokens, 16..31 <- tile1
#pragma unroll
        for (int dt = 0; dt < 4; ++dt) {
            const __bf16* vp = Vbh + (size_t)(dt * 16 + ln) * kS + jb;
            v8bf vlo = *(const v8bf*)vp;
            v8bf vhi = *(const v8bf*)(vp + 8);
            v16bf bv = __builtin_shufflevector(vlo, vhi,
                        0,1,2,3,4,5,6,7,8,9,10,11,12,13,14,15);
            acc[dt] = WMMA_BF16(af, bv, acc[dt]);
        }
    }

    // store ctx (token-major bf16 [4096, 1024], column = head*64 + d)
    const int b = bh >> 4, h = bh & 15;
#pragma unroll
    for (int r = 0; r < 8; ++r) {
        const size_t tok = (size_t)b * kS + q0 + r + 8 * hi;
#pragma unroll
        for (int dt = 0; dt < 4; ++dt)
            ctx[tok * kD + h * kHD + dt * 16 + ln] = (__bf16)acc[dt][r];
    }
}

// ---------------------------------------------------------------------------
extern "C" void kernel_launch(void* const* d_in, const int* in_sizes, int n_in,
                              void* d_out, int out_size, void* d_ws, size_t ws_size,
                              hipStream_t stream) {
    const float* query = (const float*)d_in[0];
    const float* key   = (const float*)d_in[1];
    const float* value = (const float*)d_in[2];
    const float* Wq = (const float*)d_in[3];
    const float* bq = (const float*)d_in[4];
    const float* Wk = (const float*)d_in[5];
    const float* bk = (const float*)d_in[6];
    const float* Wv = (const float*)d_in[7];
    const float* bv = (const float*)d_in[8];
    const float* Wo = (const float*)d_in[9];
    const float* bo = (const float*)d_in[10];

    float* out_p  = (float*)d_out;                 // [2,2048,1024]
    float* attn_p = out_p + kOutElems;             // [2,16,2048,2048]

    // workspace layout (bf16), 64 MB total
    char* w = (char*)d_ws;
    const size_t WSZ = (size_t)kD * kD * 2;        // 2 MB per weight
    const size_t XSZ = (size_t)kTok * kD * 2;      // 8 MB per activation
    __bf16* wq = (__bf16*)(w);
    __bf16* wk = (__bf16*)(w + WSZ);
    __bf16* wv = (__bf16*)(w + 2 * WSZ);
    __bf16* wo = (__bf16*)(w + 3 * WSZ);
    __bf16* xq = (__bf16*)(w + 4 * WSZ);
    __bf16* xk = (__bf16*)(w + 4 * WSZ + XSZ);
    __bf16* xv = (__bf16*)(w + 4 * WSZ + 2 * XSZ);
    __bf16* Qp = (__bf16*)(w + 4 * WSZ + 3 * XSZ);
    __bf16* Kp = (__bf16*)(w + 4 * WSZ + 4 * XSZ);
    __bf16* Vp = (__bf16*)(w + 4 * WSZ + 5 * XSZ);  // d-major [B,H,HD,S]
    __bf16* Cx = (__bf16*)(w + 4 * WSZ + 6 * XSZ);

    // 1) fp32 -> bf16 (vectorized)
    cvt_f32_to_bf16_v4<<<1024, 256, 0, stream>>>((const v4f*)query, (v4bf*)xq, kTok * kD / 4);
    cvt_f32_to_bf16_v4<<<1024, 256, 0, stream>>>((const v4f*)key,   (v4bf*)xk, kTok * kD / 4);
    cvt_f32_to_bf16_v4<<<1024, 256, 0, stream>>>((const v4f*)value, (v4bf*)xv, kTok * kD / 4);
    cvt_f32_to_bf16_v4<<<512, 256, 0, stream>>>((const v4f*)Wq, (v4bf*)wq, kD * kD / 4);
    cvt_f32_to_bf16_v4<<<512, 256, 0, stream>>>((const v4f*)Wk, (v4bf*)wk, kD * kD / 4);
    cvt_f32_to_bf16_v4<<<512, 256, 0, stream>>>((const v4f*)Wv, (v4bf*)wv, kD * kD / 4);
    cvt_f32_to_bf16_v4<<<512, 256, 0, stream>>>((const v4f*)Wo, (v4bf*)wo, kD * kD / 4);

    // 2) QKV projections: 256*64 tiles, 8 waves/block -> 2048 blocks
    gemm_xwT<<<2048, 256, 0, stream>>>(xq, wq, bq, Qp, kTok, kD, kD, 0);
    gemm_xwT<<<2048, 256, 0, stream>>>(xk, wk, bk, Kp, kTok, kD, kD, 0);
    gemm_xwT<<<2048, 256, 0, stream>>>(xv, wv, bv, Vp, kTok, kD, kD, 2);

    // 3) zero-fill dense attn output (bulk of HBM traffic: 537 MB, NT stores)
    zero_f32v4<<<4096, 256, 0, stream>>>((v4f*)attn_p, kAttnElems / 4);

    // 4) sparse attention: 4096 waves (2*16*128), 4 waves/block
    attn_sparse<<<1024, 128, 0, stream>>>(Qp, Kp, Vp, attn_p, Cx);

    // 5) output projection -> fp32 out
    gemm_xwT<<<2048, 256, 0, stream>>>(Cx, wo, bo, out_p, kTok, kD, kD, 1);
}
